// GAT_Cora_16045997818079
// MI455X (gfx1250) — matrix operand order repaired and checked by reading, not compiled, not run on hardware
//
#include <hip/hip_runtime.h>
#include <hip/hip_bf16.h>
#include <math.h>

typedef __attribute__((ext_vector_type(16))) _Float16 v16h;
typedef __attribute__((ext_vector_type(8)))  _Float16 v8h;
typedef __attribute__((ext_vector_type(8)))  float    v8f;

#define NN    4096
#define FIN   512
#define NH    8
#define DD    64
#define SLOPE 0.2f
#define WORDS (NN / 32)   // 128 mask words per row

// ---- helpers -----------------------------------------------------------

// A-operand (16x32 f16): per ISA, lane<16 holds K={0..7,16..23}, lane>=16 holds
// K={8..15,24..31}; elements 0..7 = first chunk, 8..15 = chunk at +16.
__device__ inline v16h ld16h_split(const _Float16* __restrict__ p) {
    union { v16h v; v8h h[2]; } u;
    u.h[0] = *(const v8h*)(p);
    u.h[1] = *(const v8h*)(p + 16);
    return u.v;
}

// Contiguous 16-half load p[0..15] (unused; B operands use ld16h_split, same K order as A).
__device__ inline v16h ld16h(const _Float16* __restrict__ p) {
    union { v16h v; v8h h[2]; } u;
    u.h[0] = *(const v8h*)(p);
    u.h[1] = *(const v8h*)(p + 8);
    return u.v;
}

__device__ inline float leaky(float e) { return e > 0.f ? e : SLOPE * e; }

// ---- prep kernels ------------------------------------------------------

__global__ void cvt_x_kernel(const float* __restrict__ x, _Float16* __restrict__ xh) {
    int i = blockIdx.x * 256 + threadIdx.x;          // N*FIN total
    xh[i] = (_Float16)x[i];
}

// W[H][FIN][DD] (f32) -> Wt[H][DD][FIN] (f16)
__global__ void cvt_w_kernel(const float* __restrict__ W, _Float16* __restrict__ Wt) {
    int idx = blockIdx.x * 256 + threadIdx.x;        // H*FIN*DD total
    int h = idx / (FIN * DD);
    int r = idx - h * FIN * DD;
    int f = r / DD;
    int d = r - f * DD;
    Wt[(h * DD + d) * FIN + f] = (_Float16)W[idx];
}

// adj int32 [N][N] -> bitmask [N][WORDS]
__global__ void bits_kernel(const int* __restrict__ adj, unsigned* __restrict__ bits) {
    int idx = blockIdx.x * 256 + threadIdx.x;        // N*WORDS total
    int i = idx / WORDS;
    int w = idx - i * WORDS;
    const int* row = adj + (size_t)i * NN + w * 32;
    unsigned mk = 0u;
    #pragma unroll
    for (int c = 0; c < 32; ++c)
        mk |= (row[c] > 0 ? 1u : 0u) << c;
    bits[idx] = mk;
}

// ---- projection: WhT[h][d][i] = sum_f x[i][f] * W[h][f][d]  (WMMA f16) ---

__global__ __launch_bounds__(32)
void proj_wmma_kernel(const _Float16* __restrict__ xh,
                      const _Float16* __restrict__ Wt,
                      _Float16* __restrict__ WhT) {
    int tile = blockIdx.x;            // 0..2047 = h*256 + row-tile
    int h    = tile >> 8;
    int mt   = tile & 255;
    int lane = threadIdx.x;
    int r15  = lane & 15;
    int aoff = (lane >> 4) << 3;      // 0 or 8  (A interleave)
    int koff = (lane >> 4) << 4;      // 0 or 16 (unused)

    const _Float16* arow  = xh + (size_t)((mt << 4) + r15) * FIN;
    const _Float16* bbase = Wt + (size_t)h * DD * FIN;

    v8f acc[4] = {v8f{}, v8f{}, v8f{}, v8f{}};

    for (int kb = 0; kb < FIN; kb += 32) {
        v16h a = ld16h_split(arow + kb + aoff);
        #pragma unroll
        for (int tI = 0; tI < 4; ++tI) {
            v16h b = ld16h_split(bbase + (size_t)(tI * 16 + r15) * FIN + kb + aoff);
            acc[tI] = __builtin_amdgcn_wmma_f32_16x16x32_f16(
                false, a, false, b, (short)0, acc[tI], false, false);
        }
    }

    // C layout: lane holds column N=r15, rows m0..m0+7 -> contiguous in WhT[d][*]
    int m0 = (mt << 4) + ((lane >> 4) << 3);
    #pragma unroll
    for (int tI = 0; tI < 4; ++tI) {
        int d = h * DD + tI * 16 + r15;
        v8h hv;
        #pragma unroll
        for (int r = 0; r < 8; ++r) hv[r] = (_Float16)acc[tI][r];
        *(v8h*)(WhT + (size_t)d * NN + m0) = hv;
    }
}

// ---- s,t scores: s[h][i] = sum_d WhT[h][d][i] * a_src[h][d] --------------

__global__ void st_kernel(const _Float16* __restrict__ WhT,
                          const float* __restrict__ a_src,
                          const float* __restrict__ a_dst,
                          float* __restrict__ s, float* __restrict__ t) {
    int i = blockIdx.x * 256 + threadIdx.x;   // node
    int h = blockIdx.y;
    float ss = 0.f, tt = 0.f;
    for (int d = 0; d < DD; ++d) {
        float w = (float)WhT[(size_t)(h * DD + d) * NN + i];
        ss += w * a_src[h * DD + d];
        tt += w * a_dst[h * DD + d];
    }
    s[h * NN + i] = ss;
    t[h * NN + i] = tt;
}

// ---- exact row max: m[h][i] = LR(s_hi + max_{j in adj(i)} t_hj) ----------

__global__ __launch_bounds__(128)
void rowmax_kernel(const unsigned* __restrict__ bits,
                   const float* __restrict__ s, const float* __restrict__ t,
                   float* __restrict__ m) {
    __shared__ float red[NH * 128];
    int i = blockIdx.x;
    int w = threadIdx.x;                       // one mask word each (WORDS==128)
    unsigned mk = bits[i * WORDS + w];
    float mx[NH];
    #pragma unroll
    for (int h = 0; h < NH; ++h) mx[h] = -INFINITY;
    while (mk) {
        int c = __ffs(mk) - 1;
        mk &= mk - 1;
        int j = w * 32 + c;
        #pragma unroll
        for (int h = 0; h < NH; ++h) mx[h] = fmaxf(mx[h], t[h * NN + j]);
    }
    #pragma unroll
    for (int h = 0; h < NH; ++h) red[h * 128 + w] = mx[h];
    __syncthreads();
    for (int sft = 64; sft > 0; sft >>= 1) {
        if (w < sft)
            #pragma unroll
            for (int h = 0; h < NH; ++h)
                red[h * 128 + w] = fmaxf(red[h * 128 + w], red[h * 128 + w + sft]);
        __syncthreads();
    }
    if (w < NH) {
        float e = leaky(s[w * NN + i] + red[w * 128]);
        if (!(e > -1e30f)) e = 0.f;            // empty-row guard
        m[w * NN + i] = e;
    }
}

// ---- per-row / per-col softmax coefficients ------------------------------
// p_ij = (s_i+t_j > 0) ? A_i*B_j : C_i*D_j   (exact split of exp(LR(.)-m))

__global__ void coef_kernel(const float* __restrict__ s, const float* __restrict__ t,
                            const float* __restrict__ m,
                            float* __restrict__ Ar, float* __restrict__ Cr,
                            float* __restrict__ Bc, float* __restrict__ Dc) {
    int idx = blockIdx.x * 256 + threadIdx.x;  // H*N total
    float sv = s[idx], tv = t[idx], mv = m[idx];
    Ar[idx] = __expf(sv - mv);
    Cr[idx] = __expf(SLOPE * sv - mv);
    Bc[idx] = __expf(tv);
    Dc[idx] = __expf(SLOPE * tv);
}

// ---- fused masked-softmax + aggregation + ELU (WMMA f16) -----------------
// one wave per (head, 32-row tile). No transcendentals in the hot loop; the
// softmax denominator comes from an extra WMMA against an all-ones B operand.

__global__ __launch_bounds__(32)
void agg_wmma_kernel(const unsigned* __restrict__ bits,
                     const float* __restrict__ sArr, const float* __restrict__ tArr,
                     const float* __restrict__ Ar, const float* __restrict__ Cr,
                     const float* __restrict__ Bc, const float* __restrict__ Dc,
                     const _Float16* __restrict__ WhT,
                     float* __restrict__ out) {
    int mt   = blockIdx.x;            // 32-row tile (0..127)
    int h    = blockIdx.y;            // head
    int lane = threadIdx.x;
    int r15  = lane & 15;
    int aoff = (lane >> 4) << 3;      // 0 or 8  (A interleave)
    int koff = (lane >> 4) << 4;      // 0 or 16 (prefetch offset)
    int row0 = (mt << 5) + r15;
    int row1 = row0 + 16;

    float sv0 = sArr[h * NN + row0], sv1 = sArr[h * NN + row1];
    float av0 = Ar[h * NN + row0],   av1 = Ar[h * NN + row1];
    float cv0 = Cr[h * NN + row0],   cv1 = Cr[h * NN + row1];

    const unsigned*  br0   = bits + (size_t)row0 * WORDS;
    const unsigned*  br1   = bits + (size_t)row1 * WORDS;
    const float*     th    = tArr + (size_t)h * NN;
    const float*     Bh    = Bc   + (size_t)h * NN;
    const float*     Dh    = Dc   + (size_t)h * NN;
    const _Float16*  bbase = WhT + (size_t)h * DD * NN;

    v8f acc0[4] = {v8f{}, v8f{}, v8f{}, v8f{}};
    v8f acc1[4] = {v8f{}, v8f{}, v8f{}, v8f{}};
    v8f den0 = {}, den1 = {};

    v16h ones;
    #pragma unroll
    for (int e = 0; e < 16; ++e) ones[e] = (_Float16)1.0f;

    for (int jb = 0; jb < NN; jb += 32) {
        unsigned mk0 = br0[jb >> 5];
        unsigned mk1 = br1[jb >> 5];

        union { v16h v; _Float16 e[16]; } a0, a1;
        #pragma unroll
        for (int c = 0; c < 16; ++c) {
            int col = aoff + (c < 8 ? c : c + 8);   // K = {aoff+0..7, aoff+16..23}
            float tv = th[jb + col];
            float bv = Bh[jb + col];
            float dv = Dh[jb + col];

            bool pos0 = (sv0 + tv) > 0.f;
            float p0  = (pos0 ? av0 : cv0) * (pos0 ? bv : dv);
            p0 = ((mk0 >> col) & 1u) ? p0 : 0.f;
            a0.e[c] = (_Float16)p0;

            bool pos1 = (sv1 + tv) > 0.f;
            float p1  = (pos1 ? av1 : cv1) * (pos1 ? bv : dv);
            p1 = ((mk1 >> col) & 1u) ? p1 : 0.f;
            a1.e[c] = (_Float16)p1;
        }

        #pragma unroll
        for (int tI = 0; tI < 4; ++tI) {
            v16h b = ld16h_split(bbase + (size_t)(tI * 16 + r15) * NN + jb + aoff);
            acc0[tI] = __builtin_amdgcn_wmma_f32_16x16x32_f16(
                false, a0.v, false, b, (short)0, acc0[tI], false, false);
            acc1[tI] = __builtin_amdgcn_wmma_f32_16x16x32_f16(
                false, a1.v, false, b, (short)0, acc1[tI], false, false);
        }
        // denominator = P x 1  (every column of the tile equals the row sum)
        den0 = __builtin_amdgcn_wmma_f32_16x16x32_f16(
            false, a0.v, false, ones, (short)0, den0, false, false);
        den1 = __builtin_amdgcn_wmma_f32_16x16x32_f16(
            false, a1.v, false, ones, (short)0, den1, false, false);

        // prefetch next B panel (L2-resident after first head pass)
        if (jb + 32 < NN)
            __builtin_prefetch((const void*)(bbase + (size_t)r15 * NN + jb + 32 + koff), 0, 1);
    }

    // C layout: lane holds column dcol, rows lrow..lrow+7 (local); den[r] is
    // already the row sum for exactly that row.
    #pragma unroll
    for (int tI = 0; tI < 4; ++tI) {
        int dcol = tI * 16 + r15;
        #pragma unroll
        for (int r = 0; r < 8; ++r) {
            int lrow = ((lane >> 4) << 3) + r;
            float v0 = acc0[tI][r] / den0[r];
            v0 = v0 > 0.f ? v0 : (__expf(v0) - 1.f);            // ELU
            out[(size_t)((mt << 5) + lrow) * (NH * DD) + h * DD + dcol] = v0;
            float v1 = acc1[tI][r] / den1[r];
            v1 = v1 > 0.f ? v1 : (__expf(v1) - 1.f);            // ELU
            out[(size_t)((mt << 5) + 16 + lrow) * (NH * DD) + h * DD + dcol] = v1;
        }
    }
}

// ---- launcher ------------------------------------------------------------

extern "C" void kernel_launch(void* const* d_in, const int* in_sizes, int n_in,
                              void* d_out, int out_size, void* d_ws, size_t ws_size,
                              hipStream_t stream) {
    const float* x     = (const float*)d_in[0];
    const int*   adj   = (const int*)d_in[1];
    const float* W     = (const float*)d_in[2];
    const float* a_src = (const float*)d_in[3];
    const float* a_dst = (const float*)d_in[4];
    float* out = (float*)d_out;

    // workspace carve-up (~12 MB total, all offsets 256B-aligned)
    char* ws = (char*)d_ws;
    _Float16* xh   = (_Float16*)ws;  ws += (size_t)NN * FIN * 2;      // 4 MB
    _Float16* Wt   = (_Float16*)ws;  ws += (size_t)NH * DD * FIN * 2; // 512 KB
    _Float16* WhT  = (_Float16*)ws;  ws += (size_t)NH * DD * NN * 2;  // 4 MB
    unsigned* bits = (unsigned*)ws;  ws += (size_t)NN * WORDS * 4;    // 2 MB
    float*    sArr = (float*)ws;     ws += (size_t)NH * NN * 4;
    float*    tArr = (float*)ws;     ws += (size_t)NH * NN * 4;
    float*    mArr = (float*)ws;     ws += (size_t)NH * NN * 4;
    float*    Ar   = (float*)ws;     ws += (size_t)NH * NN * 4;
    float*    Cr   = (float*)ws;     ws += (size_t)NH * NN * 4;
    float*    Bc   = (float*)ws;     ws += (size_t)NH * NN * 4;
    float*    Dc   = (float*)ws;     ws += (size_t)NH * NN * 4;

    cvt_x_kernel<<<(NN * FIN) / 256, 256, 0, stream>>>(x, xh);
    cvt_w_kernel<<<(NH * FIN * DD) / 256, 256, 0, stream>>>(W, Wt);
    bits_kernel<<<(NN * WORDS) / 256, 256, 0, stream>>>(adj, bits);

    proj_wmma_kernel<<<NH * (NN / 16), 32, 0, stream>>>(xh, Wt, WhT);

    st_kernel<<<dim3(NN / 256, NH), 256, 0, stream>>>(WhT, a_src, a_dst, sArr, tArr);

    rowmax_kernel<<<NN, 128, 0, stream>>>(bits, sArr, tArr, mArr);

    coef_kernel<<<(NH * NN) / 256, 256, 0, stream>>>(sArr, tArr, mArr, Ar, Cr, Bc, Dc);

    agg_wmma_kernel<<<dim3(NN / 32, NH), 32, 0, stream>>>(
        bits, sArr, tArr, Ar, Cr, Bc, Dc, WhT, out);
}
